// sLSTMEncoder_45921790328951
// MI455X (gfx1250) — matrix operand order repaired and checked by reading, not compiled, not run on hardware
//
#include <hip/hip_runtime.h>

// sLSTM: B=64, S=2048, D=128, H=128.
// Persistent-workgroup fused kernel: 4 WGs x 256 threads (8 wave32).
// Each WG owns 16 batch rows for the whole sequence; each wave owns a
// 16-column slice of H for all 4 gates, keeping c/n state in VGPRs.
// Matmuls use v_wmma_f32_16x16x32_f16; weights are pre-swizzled into LDS
// in B-fragment order; x/h tiles are double-buffered in LDS.
// Gate nonlinearities use CDNA5 hardware transcendentals (v_tanh_f32,
// v_exp_f32, v_rcp_f32) to shorten the serial per-step tail.

#define B_ 64
#define S_ 2048
#define D_ 128
#define H_ 128

typedef __attribute__((ext_vector_type(16))) _Float16 v16h;
typedef __attribute__((ext_vector_type(8)))  float    v8f;

__device__ __forceinline__ float fast_tanh(float x) {
#if __has_builtin(__builtin_amdgcn_tanhf)
    return __builtin_amdgcn_tanhf(x);
#else
    float r;
    asm("v_tanh_f32 %0, %1" : "=v"(r) : "v"(x));
    return r;
#endif
}

__device__ __forceinline__ float fast_sigmoid(float x) {
    // sigmoid(x) = 0.5*tanh(x/2) + 0.5   (1 trans + 2 fma)
    return fmaf(fast_tanh(x * 0.5f), 0.5f, 0.5f);
}

__device__ __forceinline__ float fast_exp(float x) {
    return __builtin_amdgcn_exp2f(x * 1.4426950408889634f);  // v_exp_f32
}

__device__ __forceinline__ float fast_rcp(float x) {
    return __builtin_amdgcn_rcpf(x);                          // v_rcp_f32
}

__global__ __launch_bounds__(256) void slstm_persistent_kernel(
    const float* __restrict__ x,
    const float* __restrict__ Wz, const float* __restrict__ Wi,
    const float* __restrict__ Wf, const float* __restrict__ Wo,
    const float* __restrict__ Rz, const float* __restrict__ Ri,
    const float* __restrict__ Rf, const float* __restrict__ Ro,
    const float* __restrict__ bz, const float* __restrict__ bi,
    const float* __restrict__ bf, const float* __restrict__ bo,
    float* __restrict__ out)
{
    // Weight fragments: [gate 4][nchunk 8][kchunk 8][lane 32][half 16] f16 = 256 KiB
    __shared__ __attribute__((aligned(32))) _Float16 wfrag[4 * 8 * 8 * 32 * 16];
    // x and h tiles, double buffered, rows padded to 132 halves (bank-conflict avoidance)
    __shared__ __attribute__((aligned(32))) _Float16 xbuf[2][16][132];
    __shared__ __attribute__((aligned(32))) _Float16 hbuf[2][16][132];

    const int tid  = threadIdx.x;
    const int wave = tid >> 5;           // 0..7 -> column chunk of H
    const int lane = tid & 31;
    const int nc   = wave;
    const int b0   = blockIdx.x * 16;    // batch tile base

    const float* Wp[4] = {Wz, Wi, Wf, Wo};
    const float* Rp[4] = {Rz, Ri, Rf, Ro};

    // ---- one-time prologue: swizzle weights into WMMA B-fragment order ----
    // B[k,n] with k = concat(D of W, H of R), n = output column.
    // Fragment element: lane L, half j  ->  k_local = j + 16*(L>=16), n = nc*16 + (L&15)
    for (int idx = tid; idx < 4 * 8 * 8 * 32 * 16; idx += 256) {
        int j   = idx & 15;
        int L   = (idx >> 4) & 31;
        int kk  = (idx >> 9) & 7;
        int ncc = (idx >> 12) & 7;
        int g   = (idx >> 15) & 3;
        int k   = kk * 32 + (j + ((L >> 4) << 4));
        int n   = ncc * 16 + (L & 15);
        float wv = (k < D_) ? Wp[g][n * D_ + k] : Rp[g][n * H_ + (k - D_)];
        wfrag[idx] = (_Float16)wv;
    }
    // h_{-1} = 0
    for (int idx = tid; idx < 2 * 16 * 132; idx += 256)
        ((_Float16*)hbuf)[idx] = (_Float16)0.0f;

    const int xr_row = tid >> 4;          // 0..15  batch row handled for x staging
    const int xr_d   = (tid & 15) * 8;    // 0..120 d-chunk handled for x staging
    const float4* __restrict__ xrow =
        (const float4*)&x[(size_t)(b0 + xr_row) * (S_ * D_) + xr_d];   // step stride: D_/4

    // stage x(s=0)
    {
        float4 a = xrow[0], b = xrow[1];
        _Float16* xd = &xbuf[0][xr_row][xr_d];
        xd[0] = (_Float16)a.x; xd[1] = (_Float16)a.y; xd[2] = (_Float16)a.z; xd[3] = (_Float16)a.w;
        xd[4] = (_Float16)b.x; xd[5] = (_Float16)b.y; xd[6] = (_Float16)b.z; xd[7] = (_Float16)b.w;
    }

    // per-lane constants & persistent state (C/D layout: lane L, vgpr v -> m=v+8*(L/16), n=L&15)
    const int col = nc * 16 + (lane & 15);             // hidden column owned by this lane
    const int hi8 = (lane >> 4) << 3;                  // row offset 0 or 8
    const float biasv[4] = {bz[col], bi[col], bf[col], bo[col]};
    float cst[8], nst[8];
#pragma unroll
    for (int v = 0; v < 8; ++v) { cst[v] = 0.0f; nst[v] = 1.0f; }

    // running output pointer: element (v, s) lives at o0 + v*S*H + s*H
    float* __restrict__ o0 = out + (size_t)(b0 + hi8) * (S_ * H_) + col;

    __syncthreads();

    for (int s = 0; s < S_; ++s) {
        const int buf = s & 1;

        // (1) prefetch x(s+1) into registers (overlaps with the WMMA chain)
        float4 xa{}, xb{};
        if (s + 1 < S_) {
            const float4* px = xrow + (size_t)(s + 1) * (D_ / 4);
            xa = px[0]; xb = px[1];
        }

        // (2) build A fragments: A = [x_s | h_{s-1}], 16 rows x 256 K, 8 frags of K=32.
        // A-layout used here (contiguous k, same order as wfrag B): lane m=L&15; halves j -> K = j + 16*(L>=16)  (within 32-block)
        v16h afrag[8];
#pragma unroll
        for (int kk = 0; kk < 8; ++kk) {
            const _Float16* rowp = (kk < 4) ? &xbuf[buf][lane & 15][kk * 32]
                                            : &hbuf[buf][lane & 15][(kk - 4) * 32];
            v16h a;
#pragma unroll
            for (int p = 0; p < 8; ++p) {
                int K = 2 * p + 2 * hi8;
                a[2 * p]     = rowp[K];
                a[2 * p + 1] = rowp[K + 1];
            }
            afrag[kk] = a;
        }

        // (3) four gate GEMMs: acc_g = bias_g + A @ B_g   (32 WMMAs per wave per step)
        v8f acc[4];
#pragma unroll
        for (int g = 0; g < 4; ++g) {
            v8f cacc;
#pragma unroll
            for (int v = 0; v < 8; ++v) cacc[v] = biasv[g];
#pragma unroll
            for (int kk = 0; kk < 8; ++kk) {
                const v16h bfrg = *(const v16h*)
                    &wfrag[((((g * 8 + nc) * 8 + kk) * 32) + lane) * 16];
                cacc = __builtin_amdgcn_wmma_f32_16x16x32_f16(
                    false, afrag[kk], false, bfrg, (short)0, cacc, false, false);
            }
            acc[g] = cacc;
        }

        // (4) gate nonlinearities + state update (state lives in VGPRs).
        // Hardware trans ops keep the serial tail short: tanh / exp / rcp.
        float* op = o0 + (size_t)s * H_;
#pragma unroll
        for (int v = 0; v < 8; ++v) {
            float z  = fast_tanh(acc[0][v]);
            float ig = fminf(fast_exp(acc[1][v]), 50.0f);
            float fg = fast_sigmoid(acc[2][v]);
            float og = fast_sigmoid(acc[3][v]);
            cst[v] = fmaf(fg, cst[v], ig * z);
            nst[v] = fmaf(fg, nst[v], ig);
            float h = og * (cst[v] * fast_rcp(nst[v] + 1e-6f));
            op[(size_t)v * (S_ * H_)] = h;
            hbuf[buf ^ 1][v + hi8][col] = (_Float16)h;   // h for step s+1
        }

        // (5) stage x(s+1) into the other buffer
        if (s + 1 < S_) {
            _Float16* xd = &xbuf[buf ^ 1][xr_row][xr_d];
            xd[0] = (_Float16)xa.x; xd[1] = (_Float16)xa.y; xd[2] = (_Float16)xa.z; xd[3] = (_Float16)xa.w;
            xd[4] = (_Float16)xb.x; xd[5] = (_Float16)xb.y; xd[6] = (_Float16)xb.z; xd[7] = (_Float16)xb.w;
        }

        __syncthreads();   // publish hbuf/xbuf for step s+1
    }
}

extern "C" void kernel_launch(void* const* d_in, const int* in_sizes, int n_in,
                              void* d_out, int out_size, void* d_ws, size_t ws_size,
                              hipStream_t stream) {
    (void)in_sizes; (void)n_in; (void)out_size; (void)d_ws; (void)ws_size;
    const float* x  = (const float*)d_in[0];
    const float* Wz = (const float*)d_in[1];
    const float* Wi = (const float*)d_in[2];
    const float* Wf = (const float*)d_in[3];
    const float* Wo = (const float*)d_in[4];
    const float* Rz = (const float*)d_in[5];
    const float* Ri = (const float*)d_in[6];
    const float* Rf = (const float*)d_in[7];
    const float* Ro = (const float*)d_in[8];
    const float* bz = (const float*)d_in[9];
    const float* bi = (const float*)d_in[10];
    const float* bf = (const float*)d_in[11];
    const float* bo = (const float*)d_in[12];
    float* out = (float*)d_out;

    slstm_persistent_kernel<<<dim3(B_ / 16), dim3(256), 0, stream>>>(
        x, Wz, Wi, Wf, Wo, Rz, Ri, Rf, Ro, bz, bi, bf, bo, out);
}
